// double_CrissCrossAttention_32908039422520
// MI455X (gfx1250) — matrix operand order repaired and checked by reading, not compiled, not run on hardware
//
#include <hip/hip_runtime.h>
#include <hip/hip_bf16.h>

typedef unsigned short u16;
typedef __attribute__((ext_vector_type(16))) __bf16 v16bf;
typedef __attribute__((ext_vector_type(8)))  float  v8f;

#define B_   8
#define C_   256
#define HW_  9216
#define NEGV (-1e30f)

__device__ __forceinline__ u16 f2bf(float f) {
  union { float f; unsigned int u; } v; v.f = f;
  unsigned int r = v.u + 0x7FFFu + ((v.u >> 16) & 1u);
  return (u16)(r >> 16);
}

__device__ __forceinline__ v8f wmma_bf16(v16bf a, v16bf b, v8f c) {
  // D = A(16x32 bf16) * B(32x16 bf16) + C(16x16 f32)
  return __builtin_amdgcn_wmma_f32_16x16x32_bf16(false, a, false, b, (short)0, c,
                                                 false, false);
}

// A fragment, tile A[row][k] row-major (ld elems).
// 16-bit A 16x32 layout: lane half 0 -> K {0..7,16..23}, half 1 -> K {8..15,24..31}
// Two contiguous 16B runs -> 2x b128 LDS loads.
__device__ __forceinline__ v16bf frag_a(const u16* base, int row, int ld, int half) {
  union { uint4 q[2]; v16bf v; } u;
  const u16* p = base + row * ld + half * 8;
  u.q[0] = *reinterpret_cast<const uint4*>(p);
  u.q[1] = *reinterpret_cast<const uint4*>(p + 16);
  return u.v;
}

// B fragment from TRANSPOSED tile BT[col][k] (ld elems).
// 16-bit B 32x16 layout: lane half 0 -> K {0..7,16..23}, half 1 -> K {8..15,24..31}
// Two contiguous 16B runs -> 2x b128 LDS loads.
__device__ __forceinline__ v16bf frag_bt(const u16* base, int col, int ld, int half) {
  union { uint4 q[2]; v16bf v; } u;
  const u16* p = base + col * ld + half * 8;
  u.q[0] = *reinterpret_cast<const uint4*>(p);
  u.q[1] = *reinterpret_cast<const uint4*>(p + 16);
  return u.v;
}

__device__ __forceinline__ float wave_max(float v) {
#pragma unroll
  for (int o = 16; o; o >>= 1) v = fmaxf(v, __shfl_xor(v, o, 32));
  return v;
}
__device__ __forceinline__ float wave_sum(float v) {
#pragma unroll
  for (int o = 16; o; o >>= 1) v += __shfl_xor(v, o, 32);
  return v;
}

// ---------------------------------------------------------------- weights->bf16
__global__ __launch_bounds__(256) void k_convw(const float* Wq, const float* Wk,
                                               const float* Wv, const float* Wp,
                                               u16* Wqkvb, u16* Wpb) {
  int idx = blockIdx.x * 256 + threadIdx.x;
  const int NQKV = 320 * 256;
  if (idx < NQKV) {
    int o = idx >> 8, c = idx & 255;
    float s = (o < 32) ? Wq[o * 256 + c]
              : (o < 64) ? Wk[(o - 32) * 256 + c] : Wv[(o - 64) * 256 + c];
    Wqkvb[idx] = f2bf(s);
  } else {
    int j = idx - NQKV;
    if (j < 256 * 512) Wpb[j] = f2bf(Wp[j]);
  }
}

// ----------------------------------------------------- fused QKV projection GEMM
// D[320,9216] = Wqkv[320,256] x src[256,9216] per (stream,batch). bf16 WMMA.
__global__ __launch_bounds__(256) void k_qkv(
    const float* x, const float* y, const u16* Wqkvb,
    const float* bq, const float* bk, const float* bv,
    u16* qT, u16* kT, u16* kcol, u16* vb, u16* vcol) {
  int sb = blockIdx.z;                 // s*8 + b
  int bat = sb & 7;
  const float* src = (sb >> 3) ? y : x;
  int n0 = blockIdx.x * 64, m0 = blockIdx.y * 64;
  __shared__ u16 Als[64 * 32];         // A[m][k]
  __shared__ u16 BlsT[64 * 32];        // B transposed: BT[n][k]
  int tid = threadIdx.x, wv = tid >> 5, lane = tid & 31;
  int half = lane >> 4, l16 = lane & 15;
  int wm = wv & 3, wn = wv >> 2;
  v8f acc0 = {}, acc1 = {};
  for (int k0 = 0; k0 < 256; k0 += 32) {
    { int r = tid >> 2, c = (tid & 3) * 8;
      *reinterpret_cast<uint4*>(&Als[r * 32 + c]) =
          *reinterpret_cast<const uint4*>(Wqkvb + (m0 + r) * 256 + k0 + c); }
    { int r = tid >> 3, c = (tid & 7) * 8;
      const float* g = src + ((size_t)bat * C_ + k0 + r) * HW_ + n0 + c;
      float4 f0 = reinterpret_cast<const float4*>(g)[0];
      float4 f1 = reinterpret_cast<const float4*>(g)[1];
      if (k0 + 32 < 256) __builtin_prefetch(g + 32 * HW_, 0, 1);
      BlsT[(c + 0) * 32 + r] = f2bf(f0.x); BlsT[(c + 1) * 32 + r] = f2bf(f0.y);
      BlsT[(c + 2) * 32 + r] = f2bf(f0.z); BlsT[(c + 3) * 32 + r] = f2bf(f0.w);
      BlsT[(c + 4) * 32 + r] = f2bf(f1.x); BlsT[(c + 5) * 32 + r] = f2bf(f1.y);
      BlsT[(c + 6) * 32 + r] = f2bf(f1.z); BlsT[(c + 7) * 32 + r] = f2bf(f1.w); }
    __syncthreads();
    v16bf a  = frag_a(Als, wm * 16 + l16, 32, half);
    v16bf b0 = frag_bt(BlsT, wn * 32 + l16, 32, half);
    v16bf b1 = frag_bt(BlsT, wn * 32 + 16 + l16, 32, half);
    acc0 = wmma_bf16(a, b0, acc0);
    acc1 = wmma_bf16(a, b1, acc1);
    __syncthreads();
  }
  int mBase = m0 + wm * 16;
#pragma unroll
  for (int hc = 0; hc < 2; ++hc) {
    v8f d = hc ? acc1 : acc0;
    int col = n0 + wn * 32 + hc * 16 + l16;
    int wcol = col % 96, hrow = col / 96;
#pragma unroll
    for (int r = 0; r < 8; ++r) {
      int mg = mBase + half * 8 + r;
      float bias = (mg < 32) ? bq[mg] : (mg < 64) ? bk[mg - 32] : bv[mg - 64];
      u16 o = f2bf(d[r] + bias);
      if (mg < 32) {                         // q, pixel-major [pix][32]
        qT[((size_t)sb * HW_ + col) * 32 + mg] = o;
      } else if (mg < 64) {                  // k, pixel-major + column-major copies
        int c = mg - 32;
        kT[((size_t)sb * HW_ + col) * 32 + c] = o;
        kcol[(((size_t)sb * 96 + wcol) * 96 + hrow) * 32 + c] = o;
      } else {                               // v, natural + column-major copies
        int c = mg - 64;
        vb[((size_t)sb * 256 + c) * HW_ + col] = o;
        vcol[(((size_t)sb * 96 + wcol) * 256 + c) * 96 + hrow] = o;
      }
    }
  }
}

// ----------------------------------------------- column logits eH = q^T k (+mask)
__global__ __launch_bounds__(256) void k_logits_h(const u16* qT, const u16* kcol,
                                                  float* logitsH) {
  int w = blockIdx.x, sb = blockIdx.y;
  __shared__ u16 Als[96 * 32];   // A[h][c]
  __shared__ u16 Bls[96 * 32];   // BT[j][c] (straight copy of kcol[w] block)
  int tid = threadIdx.x, wv = tid >> 5, lane = tid & 31;
  int half = lane >> 4, l16 = lane & 15;
  for (int i = tid; i < 96 * 4; i += 256) {
    int hh = i >> 2, c8 = (i & 3) * 8;
    *reinterpret_cast<uint4*>(&Als[hh * 32 + c8]) =
        *reinterpret_cast<const uint4*>(qT + ((size_t)sb * HW_ + hh * 96 + w) * 32 + c8);
  }
  { const uint4* s4 = reinterpret_cast<const uint4*>(kcol + ((size_t)sb * 96 + w) * 3072);
    for (int i = tid; i < 384; i += 256) reinterpret_cast<uint4*>(Bls)[i] = s4[i]; }
  __syncthreads();
  for (int t = wv; t < 36; t += 8) {
    int h0 = (t / 6) * 16, j0 = (t % 6) * 16;
    v16bf a  = frag_a(Als, h0 + l16, 32, half);
    v16bf bb = frag_bt(Bls, j0 + l16, 32, half);
    v8f d = {};
    d = wmma_bf16(a, bb, d);
    int j = j0 + l16;
#pragma unroll
    for (int r = 0; r < 8; ++r) {
      int hg = h0 + half * 8 + r;
      logitsH[(((size_t)sb * 96 + w) * 96 + hg) * 96 + j] = (hg == j) ? NEGV : d[r];
    }
  }
}

// -------------------------------- row logits eW (WMMA) + fused softmax over 192
__global__ __launch_bounds__(256) void k_logits_w_sm(const u16* qT, const u16* kT,
                                                     const float* logitsH, u16* attb) {
  int h = blockIdx.x, sb = blockIdx.y;
  __shared__ u16 Als[96 * 32];   // A[w][c]  (straight copy of qT row block)
  __shared__ u16 Bls[96 * 32];   // BT[j][c] (straight copy of kT row block)
  __shared__ float eWs[96 * 96];
  int tid = threadIdx.x, wv = tid >> 5, lane = tid & 31;
  int half = lane >> 4, l16 = lane & 15;
  { const uint4* s4 = reinterpret_cast<const uint4*>(qT + ((size_t)sb * HW_ + h * 96) * 32);
    for (int i = tid; i < 384; i += 256) reinterpret_cast<uint4*>(Als)[i] = s4[i]; }
  { const uint4* s4 = reinterpret_cast<const uint4*>(kT + ((size_t)sb * HW_ + h * 96) * 32);
    for (int i = tid; i < 384; i += 256) reinterpret_cast<uint4*>(Bls)[i] = s4[i]; }
  __syncthreads();
  for (int t = wv; t < 36; t += 8) {
    int w0 = (t / 6) * 16, j0 = (t % 6) * 16;
    v16bf a  = frag_a(Als, w0 + l16, 32, half);
    v16bf bb = frag_bt(Bls, j0 + l16, 32, half);
    v8f d = {};
    d = wmma_bf16(a, bb, d);
#pragma unroll
    for (int r = 0; r < 8; ++r) eWs[(w0 + half * 8 + r) * 96 + j0 + l16] = d[r];
  }
  __syncthreads();
  for (int wr = wv; wr < 96; wr += 8) {       // softmax per pixel (h,wr), 192 logits
    const float* eh = logitsH + (((size_t)sb * 96 + wr) * 96 + h) * 96;
    float vals[6];
#pragma unroll
    for (int i = 0; i < 3; ++i) vals[i] = eh[lane + i * 32];
#pragma unroll
    for (int i = 0; i < 3; ++i) vals[3 + i] = eWs[wr * 96 + lane + i * 32];
    float m = vals[0];
#pragma unroll
    for (int i = 1; i < 6; ++i) m = fmaxf(m, vals[i]);
    m = wave_max(m);
    float s = 0.f;
#pragma unroll
    for (int i = 0; i < 6; ++i) { vals[i] = __expf(vals[i] - m); s += vals[i]; }
    float inv = 1.f / wave_sum(s);
    u16* ab = attb + ((size_t)sb * HW_ + h * 96 + wr) * 192;
#pragma unroll
    for (int i = 0; i < 3; ++i) ab[lane + i * 32] = f2bf(vals[i] * inv);
#pragma unroll
    for (int i = 0; i < 3; ++i) ab[96 + lane + i * 32] = f2bf(vals[3 + i] * inv);
  }
}

// --------------------------------------- column aggregation outH = v_col x aH^T
// B operand BT[h][j] = aH[h][j] is attb's natural row order -> straight copy.
__global__ __launch_bounds__(256) void k_aggr_h(const u16* vcol, const u16* attb,
                                                float* z) {
  int w = blockIdx.x, sb = blockIdx.y;
  int s = sb >> 3, bat = sb & 7;
  int vsb = sb ^ 8;                    // cross-stream V (vy for stream x, vx for y)
  __shared__ u16 Avs[128 * 96];        // v[c][j] for column w (half of C at a time)
  __shared__ u16 Bas[96 * 96];         // BT[h][j] = aH
  int tid = threadIdx.x, wv = tid >> 5, lane = tid & 31;
  int half = lane >> 4, l16 = lane & 15;
  for (int i = tid; i < 96 * 12; i += 256) {
    int hh = i / 12, j8 = (i % 12) * 8;
    *reinterpret_cast<uint4*>(&Bas[hh * 96 + j8]) =
        *reinterpret_cast<const uint4*>(attb + ((size_t)sb * HW_ + hh * 96 + w) * 192 + j8);
  }
  for (int cp = 0; cp < 2; ++cp) {
    __syncthreads();
    const uint4* s4 = reinterpret_cast<const uint4*>(
        vcol + ((size_t)vsb * 96 + w) * 24576 + cp * 12288);
    for (int i = tid; i < 1536; i += 256) reinterpret_cast<uint4*>(Avs)[i] = s4[i];
    __syncthreads();
    for (int t = wv; t < 48; t += 8) {
      int lc0 = (t / 6) * 16, h0 = (t % 6) * 16;
      v8f d = {};
#pragma unroll
      for (int k0 = 0; k0 < 96; k0 += 32) {
        v16bf a  = frag_a(Avs + k0, lc0 + l16, 96, half);
        v16bf bb = frag_bt(Bas + k0, h0 + l16, 96, half);
        d = wmma_bf16(a, bb, d);
      }
#pragma unroll
      for (int r = 0; r < 8; ++r) {
        int c = cp * 128 + lc0 + half * 8 + r;
        z[((size_t)bat * 512 + s * 256 + c) * HW_ + (h0 + l16) * 96 + w] = d[r];
      }
    }
  }
}

// ---------------- row aggregation outW + epilogue z = gamma*(outH+outW)+residual
__global__ __launch_bounds__(256) void k_aggr_w(const u16* vb, const u16* attb,
                                                const float* x, const float* y,
                                                const float* g1, const float* g2,
                                                float* z) {
  int h = blockIdx.x, sb = blockIdx.y;
  int s = sb >> 3, bat = sb & 7;
  int vsb = sb ^ 8;
  const float* res = s ? y : x;
  float g = s ? g2[0] : g1[0];
  __shared__ u16 Avs[128 * 96];        // v[c][j] for row h
  __shared__ u16 Bas[96 * 96];         // BT[w][j] = aW
  int tid = threadIdx.x, wv = tid >> 5, lane = tid & 31;
  int half = lane >> 4, l16 = lane & 15;
  for (int i = tid; i < 96 * 12; i += 256) {
    int ww = i / 12, j8 = (i % 12) * 8;
    *reinterpret_cast<uint4*>(&Bas[ww * 96 + j8]) =
        *reinterpret_cast<const uint4*>(
            attb + ((size_t)sb * HW_ + h * 96 + ww) * 192 + 96 + j8);
  }
  for (int cp = 0; cp < 2; ++cp) {
    __syncthreads();
    for (int i = tid; i < 128 * 12; i += 256) {
      int lc = i / 12, j8 = (i % 12) * 8;
      *reinterpret_cast<uint4*>(&Avs[lc * 96 + j8]) =
          *reinterpret_cast<const uint4*>(
              vb + ((size_t)vsb * 256 + cp * 128 + lc) * HW_ + h * 96 + j8);
    }
    __syncthreads();
    for (int t = wv; t < 48; t += 8) {
      int lc0 = (t / 6) * 16, w0 = (t % 6) * 16;
      v8f d = {};
#pragma unroll
      for (int k0 = 0; k0 < 96; k0 += 32) {
        v16bf a  = frag_a(Avs + k0, lc0 + l16, 96, half);
        v16bf bb = frag_bt(Bas + k0, w0 + l16, 96, half);
        d = wmma_bf16(a, bb, d);
      }
#pragma unroll
      for (int r = 0; r < 8; ++r) {
        int c = cp * 128 + lc0 + half * 8 + r;
        int wc = w0 + l16;
        size_t zi = ((size_t)bat * 512 + s * 256 + c) * HW_ + h * 96 + wc;
        z[zi] = g * (z[zi] + d[r]) + res[((size_t)bat * C_ + c) * HW_ + h * 96 + wc];
      }
    }
  }
}

// ------------------------------------------------ depthwise 3x3 SAME, fp32->bf16
__global__ __launch_bounds__(256) void k_dw(const float* z, const float* Wd,
                                            const float* bd, u16* zd) {
  size_t idx = (size_t)blockIdx.x * 256 + threadIdx.x;
  int pix = (int)(idx % HW_);
  int ch  = (int)((idx / HW_) % 512);
  int bat = (int)(idx / ((size_t)HW_ * 512));
  int hh = pix / 96, ww = pix % 96;
  float acc = bd[ch];
  const float* zp = z + ((size_t)bat * 512 + ch) * HW_;
  const float* wd = Wd + ch * 9;
#pragma unroll
  for (int dy = -1; dy <= 1; ++dy) {
    int ih = hh + dy;
    if (ih < 0 || ih >= 96) continue;
#pragma unroll
    for (int dx = -1; dx <= 1; ++dx) {
      int iw = ww + dx;
      if (iw < 0 || iw >= 96) continue;
      acc += wd[(dy + 1) * 3 + (dx + 1)] * zp[ih * 96 + iw];
    }
  }
  zd[idx] = f2bf(acc);
}

// ---------------------------------------- final 1x1 projection + leaky-ReLU GEMM
__global__ __launch_bounds__(256) void k_proj(const u16* Wpb, const u16* zd,
                                              const float* bp, float* out) {
  int bat = blockIdx.z;
  int n0 = blockIdx.x * 64, m0 = blockIdx.y * 64;
  __shared__ u16 Als[64 * 32];
  __shared__ u16 BlsT[64 * 32];
  int tid = threadIdx.x, wv = tid >> 5, lane = tid & 31;
  int half = lane >> 4, l16 = lane & 15;
  int wm = wv & 3, wn = wv >> 2;
  v8f acc0 = {}, acc1 = {};
  for (int k0 = 0; k0 < 512; k0 += 32) {
    { int r = tid >> 2, c = (tid & 3) * 8;
      *reinterpret_cast<uint4*>(&Als[r * 32 + c]) =
          *reinterpret_cast<const uint4*>(Wpb + (m0 + r) * 512 + k0 + c); }
    { int r = tid >> 3, c = (tid & 7) * 8;
      const u16* g = zd + ((size_t)bat * 512 + k0 + r) * HW_ + n0 + c;
      union { uint4 v; u16 s[8]; } u;
      u.v = *reinterpret_cast<const uint4*>(g);
      if (k0 + 32 < 512) __builtin_prefetch(g + (size_t)32 * HW_, 0, 1);
#pragma unroll
      for (int t = 0; t < 8; ++t) BlsT[(c + t) * 32 + r] = u.s[t]; }
    __syncthreads();
    v16bf a  = frag_a(Als, wm * 16 + l16, 32, half);
    v16bf b0 = frag_bt(BlsT, wn * 32 + l16, 32, half);
    v16bf b1 = frag_bt(BlsT, wn * 32 + 16 + l16, 32, half);
    acc0 = wmma_bf16(a, b0, acc0);
    acc1 = wmma_bf16(a, b1, acc1);
    __syncthreads();
  }
  int mBase = m0 + wm * 16;
#pragma unroll
  for (int hc = 0; hc < 2; ++hc) {
    v8f d = hc ? acc1 : acc0;
    int col = n0 + wn * 32 + hc * 16 + l16;
#pragma unroll
    for (int r = 0; r < 8; ++r) {
      int mg = mBase + half * 8 + r;
      float v = d[r] + bp[mg];
      out[((size_t)bat * C_ + mg) * HW_ + col] = v > 0.f ? v : 0.01f * v;
    }
  }
}

extern "C" void kernel_launch(void* const* d_in, const int* in_sizes, int n_in,
                              void* d_out, int out_size, void* d_ws, size_t ws_size,
                              hipStream_t stream) {
  const float* x  = (const float*)d_in[0];
  const float* y  = (const float*)d_in[1];
  const float* Wq = (const float*)d_in[2];
  const float* bq = (const float*)d_in[3];
  const float* Wk = (const float*)d_in[4];
  const float* bk = (const float*)d_in[5];
  const float* Wv = (const float*)d_in[6];
  const float* bv = (const float*)d_in[7];
  const float* g1 = (const float*)d_in[8];
  const float* g2 = (const float*)d_in[9];
  const float* Wd = (const float*)d_in[10];
  const float* bd = (const float*)d_in[11];
  const float* Wp = (const float*)d_in[12];
  const float* bp = (const float*)d_in[13];
  float* out = (float*)d_out;

  // workspace carve-up (all offsets 16B aligned); total ~520 MB
  char* p = (char*)d_ws;
  u16* Wqkvb = (u16*)p; p += (size_t)320 * 256 * 2;
  u16* Wpb   = (u16*)p; p += (size_t)256 * 512 * 2;
  u16* qT    = (u16*)p; p += (size_t)16 * HW_ * 32 * 2;     // [sb][pix][32]
  u16* kT    = (u16*)p; p += (size_t)16 * HW_ * 32 * 2;     // [sb][pix][32]
  u16* kcol  = (u16*)p; p += (size_t)16 * 96 * 96 * 32 * 2; // [sb][w][h][c]
  u16* vb    = (u16*)p; p += (size_t)16 * 256 * HW_ * 2;    // [sb][c][pix]
  u16* vcol  = (u16*)p; p += (size_t)16 * 96 * 256 * 96 * 2;// [sb][w][c][h]
  float* logitsH = (float*)p; p += (size_t)16 * 96 * 96 * 96 * 4; // [sb][w][h][j]
  u16* attb  = (u16*)p; p += (size_t)16 * HW_ * 192 * 2;    // [sb][pix][192]
  float* z   = (float*)p; p += (size_t)B_ * 512 * HW_ * 4;
  u16* zd    = (u16*)p; p += (size_t)B_ * 512 * HW_ * 2;

  k_convw<<<dim3(832), 256, 0, stream>>>(Wq, Wk, Wv, Wp, Wqkvb, Wpb);
  k_qkv<<<dim3(144, 5, 16), 256, 0, stream>>>(x, y, Wqkvb, bq, bk, bv,
                                              qT, kT, kcol, vb, vcol);
  k_logits_h<<<dim3(96, 16), 256, 0, stream>>>(qT, kcol, logitsH);
  k_logits_w_sm<<<dim3(96, 16), 256, 0, stream>>>(qT, kT, logitsH, attb);
  k_aggr_h<<<dim3(96, 16), 256, 0, stream>>>(vcol, attb, z);
  k_aggr_w<<<dim3(96, 16), 256, 0, stream>>>(vb, attb, x, y, g1, g2, z);
  k_dw<<<dim3(147456), 256, 0, stream>>>(z, Wd, bd, zd);
  k_proj<<<dim3(144, 4, 8), 256, 0, stream>>>(Wpb, zd, bp, out);
}
